// ContrastiveLoss_28372553957605
// MI455X (gfx1250) — matrix operand order repaired and checked by reading, not compiled, not run on hardware
//
#include <hip/hip_runtime.h>
#include <hip/hip_bf16.h>
#include <math.h>

// ---- types for WMMA -------------------------------------------------------
typedef __attribute__((ext_vector_type(16))) __bf16 v16bf;
typedef __attribute__((ext_vector_type(8)))  __bf16 v8bf;
typedef __attribute__((ext_vector_type(8)))  float  v8f;

#define N_ROWS 8192
#define DDIM   512
#define BTM    256   // block tile M (8 waves: 4 in M x 2 in N, 64x64 per wave)
#define BTN    128   // block tile N
#define KT     32    // K per WMMA / per LDS stage
#define LDSS   40    // bf16 elements per LDS row (32 + 8 pad -> 80B stride)

// ---------------------------------------------------------------------------
// Kernel 1: fused L2-normalize + fp32->bf16 cast. One wave32 per row.
// rows [0,8192) come from cxr -> A, rows [8192,16384) from ehr -> B.
// ---------------------------------------------------------------------------
__global__ __launch_bounds__(256) void nrm_kernel(
    const float* __restrict__ cxr, const float* __restrict__ ehr,
    __bf16* __restrict__ A, __bf16* __restrict__ B) {
  const int wave = threadIdx.x >> 5;
  const int lane = threadIdx.x & 31;
  const int row  = blockIdx.x * 8 + wave;           // 0..16383

  const float* src;
  __bf16*      dst;
  if (row < N_ROWS) {
    src = cxr + (size_t)row * DDIM;
    dst = A   + (size_t)row * DDIM;
  } else {
    src = ehr + (size_t)(row - N_ROWS) * DDIM;
    dst = B   + (size_t)(row - N_ROWS) * DDIM;
  }

  // each lane owns 16 consecutive columns
  float vals[16];
  const float4* p = (const float4*)(src + lane * 16);
  float ss = 0.f;
#pragma unroll
  for (int j = 0; j < 4; ++j) {
    float4 t = p[j];
    vals[4 * j + 0] = t.x; vals[4 * j + 1] = t.y;
    vals[4 * j + 2] = t.z; vals[4 * j + 3] = t.w;
    ss += t.x * t.x + t.y * t.y + t.z * t.z + t.w * t.w;
  }
  // wave32 all-reduce
#pragma unroll
  for (int m = 16; m; m >>= 1) ss += __shfl_xor(ss, m, 32);
  const float scale = 1.0f / fmaxf(sqrtf(ss), 1e-8f);

  v8bf o0, o1;
#pragma unroll
  for (int i = 0; i < 8; ++i) {
    o0[i] = (__bf16)(vals[i]     * scale);
    o1[i] = (__bf16)(vals[8 + i] * scale);
  }
  *(v8bf*)(dst + lane * 16)     = o0;
  *(v8bf*)(dst + lane * 16 + 8) = o1;
}

// ---------------------------------------------------------------------------
// Kernel 2: 256x128 tile of sim = A_n @ B_n^T / tau via bf16 WMMA, fused
// epilogue: exp(sim - 1/tau) accumulated into row/col sums (diag masked),
// diagonal sims accumulated into a scalar.
// 8 waves, wave grid 4(M) x 2(N); each wave owns a 64x64 patch = 4x4 WMMA
// accumulators (raises LDS arithmetic intensity to ~33 FLOP/byte so the
// matrix pipes are not LDS-bandwidth starved).
// ---------------------------------------------------------------------------
__global__ __launch_bounds__(256, 1) void sim_kernel(
    const __bf16* __restrict__ A, const __bf16* __restrict__ B,
    const float* __restrict__ temp_ptr,
    float* __restrict__ row_sum, float* __restrict__ col_sum,
    float* __restrict__ diag_sum) {
  __shared__ __bf16 lA[BTM * LDSS];   // 20 KB
  __shared__ __bf16 lB[BTN * LDSS];   // 10 KB

  const int tid  = threadIdx.x;
  const int lane = tid & 31;
  const int wave = tid >> 5;
  const int wm   = wave >> 1;      // 0..3 : 64 rows each
  const int wn   = wave & 1;       // 0..1 : 64 cols each
  const int m0   = blockIdx.x * BTM;
  const int n0   = blockIdx.y * BTN;
  const int lrow = lane & 15;      // row-in-tile (A) or col-in-tile (B)
  const int half = lane >> 4;      // half-wave selector

  v8f acc[4][4] = {};

  for (int k0 = 0; k0 < DDIM; k0 += KT) {
    // ---- cooperative global -> LDS stage (16KB A + 8KB B per chunk) ----
#pragma unroll
    for (int it = 0; it < 4; ++it) {
      const int c  = tid + it * 256;     // 0..1023 chunk id (A: 256 rows x 4)
      const int r  = c >> 2;
      const int cc = (c & 3) * 8;        // bf16 col offset 0/8/16/24
      *(v8bf*)&lA[r * LDSS + cc] = *(const v8bf*)(A + (size_t)(m0 + r) * DDIM + k0 + cc);
    }
#pragma unroll
    for (int it = 0; it < 2; ++it) {
      const int c  = tid + it * 256;     // 0..511 chunk id (B: 128 rows x 4)
      const int r  = c >> 2;
      const int cc = (c & 3) * 8;
      *(v8bf*)&lB[r * LDSS + cc] = *(const v8bf*)(B + (size_t)(n0 + r) * DDIM + k0 + cc);
    }
    // prefetch next K chunk while we compute (global_prefetch_b8)
    if (k0 + KT < DDIM) {
      __builtin_prefetch((const void*)(A + (size_t)(m0 + tid) * DDIM + k0 + KT), 0, 3);
      __builtin_prefetch((const void*)(B + (size_t)(n0 + (tid & 127)) * DDIM + k0 + KT + ((tid >> 7) << 4)), 0, 3);
    }
    __syncthreads();

    // ---- fragment loads (ISA wave32 layouts) ----
    // A 16x32: lanes 0-15 (M=lrow) hold K{0..7,16..23}; lanes 16-31 K{8..15,24..31}
    v16bf af[4];
#pragma unroll
    for (int mi = 0; mi < 4; ++mi) {
      const __bf16* rp = &lA[(wm * 64 + mi * 16 + lrow) * LDSS];
      v8bf lo = *(const v8bf*)(rp + half * 8);
      v8bf hi = *(const v8bf*)(rp + 16 + half * 8);
      af[mi] = __builtin_shufflevector(lo, hi, 0, 1, 2, 3, 4, 5, 6, 7,
                                       8, 9, 10, 11, 12, 13, 14, 15);
    }
    // B 32x16: lanes 0-15 (N=lrow) hold K{0..7,16..23}; lanes 16-31 K{8..15,24..31}
    v16bf bfr[4];
#pragma unroll
    for (int ni = 0; ni < 4; ++ni) {
      const __bf16* rp = &lB[(wn * 64 + ni * 16 + lrow) * LDSS];
      v8bf lo = *(const v8bf*)(rp + half * 8);
      v8bf hi = *(const v8bf*)(rp + 16 + half * 8);
      bfr[ni] = __builtin_shufflevector(lo, hi, 0, 1, 2, 3, 4, 5, 6, 7,
                                        8, 9, 10, 11, 12, 13, 14, 15);
    }

    // ---- 16 WMMAs per wave per K-step ----
#pragma unroll
    for (int mi = 0; mi < 4; ++mi)
#pragma unroll
      for (int ni = 0; ni < 4; ++ni)
        acc[mi][ni] = __builtin_amdgcn_wmma_f32_16x16x32_bf16(
            false, af[mi], false, bfr[ni], (short)0, acc[mi][ni], false, false);

    __syncthreads();
  }

  // ---- fused masked-logsumexp epilogue ----
  // Fixed shift M = 1/tau is exact for logsumexp and safe since sim <= 1/tau.
  const float inv_t = 1.0f / temp_ptr[0];
#pragma unroll
  for (int mi = 0; mi < 4; ++mi) {
#pragma unroll
    for (int ni = 0; ni < 4; ++ni) {
      const int gn = n0 + wn * 64 + ni * 16 + lrow;  // this lane's column
      float rs[8];
      float csum = 0.f;
#pragma unroll
      for (int r = 0; r < 8; ++r) {
        const int gm  = m0 + wm * 64 + mi * 16 + r + half * 8;
        const float sim = acc[mi][ni][r] * inv_t;
        float e = __expf(sim - inv_t);
        if (gm == gn) {                 // diagonal: record positive, mask from lse
          atomicAdd(diag_sum, sim);
          e = 0.f;
        }
        rs[r] = e;
        csum += e;
      }
      // row sums: reduce across the 16 lanes of this half (all n, fixed m)
#pragma unroll
      for (int r = 0; r < 8; ++r) {
        float v = rs[r];
#pragma unroll
        for (int m = 8; m; m >>= 1) v += __shfl_xor(v, m, 32);
        if (lrow == 0) {
          const int gm = m0 + wm * 64 + mi * 16 + r + half * 8;
          atomicAdd(row_sum + gm, v);
        }
      }
      // col sums: combine the two half-waves (m 0..7 + m 8..15)
      const float cv = csum + __shfl_xor(csum, 16, 32);
      if (half == 0) atomicAdd(col_sum + gn, cv);
    }
  }
}

// ---------------------------------------------------------------------------
// Kernel 3: loss = (sum_i log(rs_i) + log(cs_i) + 2N/tau - 2*sum_diag) / N
// ---------------------------------------------------------------------------
__global__ __launch_bounds__(256) void fin_kernel(
    const float* __restrict__ row_sum, const float* __restrict__ col_sum,
    const float* __restrict__ diag_sum, const float* __restrict__ temp_ptr,
    float* __restrict__ out) {
  __shared__ float red[256];
  float acc = 0.f;
  for (int i = threadIdx.x; i < N_ROWS; i += 256)
    acc += logf(row_sum[i]) + logf(col_sum[i]);
  red[threadIdx.x] = acc;
  __syncthreads();
  for (int s = 128; s; s >>= 1) {
    if (threadIdx.x < s) red[threadIdx.x] += red[threadIdx.x + s];
    __syncthreads();
  }
  if (threadIdx.x == 0) {
    const float inv_t = 1.0f / temp_ptr[0];
    const float total = red[0] + 2.0f * (float)N_ROWS * inv_t - 2.0f * diag_sum[0];
    out[0] = total / (float)N_ROWS;
  }
}

// ---------------------------------------------------------------------------
extern "C" void kernel_launch(void* const* d_in, const int* in_sizes, int n_in,
                              void* d_out, int out_size, void* d_ws, size_t ws_size,
                              hipStream_t stream) {
  (void)in_sizes; (void)n_in; (void)out_size; (void)ws_size;
  const float* cxr  = (const float*)d_in[0];
  const float* ehr  = (const float*)d_in[1];
  const float* temp = (const float*)d_in[2];

  char* ws = (char*)d_ws;
  const size_t mat_bytes = (size_t)N_ROWS * DDIM * sizeof(__bf16);  // 8 MB
  __bf16* A        = (__bf16*)ws;
  __bf16* B        = (__bf16*)(ws + mat_bytes);
  float*  row_sum  = (float*)(ws + 2 * mat_bytes);
  float*  col_sum  = row_sum + N_ROWS;
  float*  diag_sum = col_sum + N_ROWS;

  // zero the accumulators every call (graph-replay safe)
  hipMemsetAsync(row_sum, 0, (2 * N_ROWS + 4) * sizeof(float), stream);

  nrm_kernel<<<2 * N_ROWS / 8, 256, 0, stream>>>(cxr, ehr, A, B);

  dim3 grid(N_ROWS / BTM, N_ROWS / BTN);
  sim_kernel<<<grid, 256, 0, stream>>>(A, B, temp, row_sum, col_sum, diag_sum);

  fin_kernel<<<1, 256, 0, stream>>>(row_sum, col_sum, diag_sum, temp, (float*)d_out);
}
